// RelationalGTEncoderLayer_26225070309507
// MI455X (gfx1250) — matrix operand order repaired and checked by reading, not compiled, not run on hardware
//
#include <hip/hip_runtime.h>
#include <hip/hip_bf16.h>
#include <math.h>

// ---------------------------------------------------------------------------
// CDNA5 (gfx1250) RelationalGT encoder layer, wave32 + WMMA bf16 + TDM.
//  - Weights pre-transposed/converted once to bf16 (N x K): both A- and
//    B-fragments are contiguous packed-pair u32 loads.
//  - GEMM: 64x16 per wave (4 A-frags x 1 shared B-frag per k-step).
//  - Flash attention: K/V tiles double-buffered into LDS by the Tensor Data
//    Mover (tensor_load_to_lds + s_wait_tensorcnt), per-relation masked P*V,
//    fragment reads via true LDS (ds_load) path.
// ---------------------------------------------------------------------------

typedef __attribute__((ext_vector_type(16))) __bf16  v16bf;
typedef __attribute__((ext_vector_type(8)))  float   v8f;
typedef __attribute__((ext_vector_type(4)))  unsigned int v4u;
typedef __attribute__((ext_vector_type(8)))  int     v8i;
typedef __attribute__((ext_vector_type(4)))  int     v4i;

union Frag {
    v16bf          v;
    unsigned int   u[8];
    unsigned short s[16];
};

__device__ __forceinline__ unsigned short f32_to_bf16_bits(float f) {
    unsigned u = __builtin_bit_cast(unsigned, f);
    unsigned r = u + 0x7FFFu + ((u >> 16) & 1u);   // round-to-nearest-even
    return (unsigned short)(r >> 16);
}

#define MODE_BF16      0
#define MODE_F32       1
#define MODE_GELU_BF16 2
#define MODE_BF16_VT   3   // bf16 transposed out: (B, Nout, 1024)

// ---------------------------------------------------------------------------
// TDM: issue a (up to 3D) bf16 tile load global -> LDS. Descriptor packing
// per CDNA5 ISA §8.3/§8.4. All arguments wave-uniform. data_size = 2 bytes.
//   dim0 = contiguous elements per line (tile_d0)
//   dim1 = lines (tile_d1), line stride = s0 elements
//   dim2 = planes (tile_d2, 0 = unused), plane stride = s1 elements
// ---------------------------------------------------------------------------
__device__ __forceinline__ void tdm_load_bf16(unsigned lds_addr,
                                              unsigned long long gaddr,
                                              unsigned tile_d0, unsigned tile_d1,
                                              unsigned long long s0,
                                              unsigned tile_d2,
                                              unsigned long long s1)
{
    const unsigned long long td0 = 1ull << 20, td1 = 1ull << 20;
    v4u g0;
    g0[0] = 1u;                                            // count=1, user mode
    g0[1] = lds_addr;                                      // LDS byte address
    g0[2] = (unsigned)gaddr;                               // global_addr[31:0]
    g0[3] = (unsigned)((gaddr >> 32) & 0x01FFFFFFull) |    // global_addr[56:32]
            (2u << 30);                                    // type = 2 (image)
    v8i g1;
    g1[0] = (int)(1u << 16);                               // data_size=1 -> 2B
    g1[1] = (int)(((unsigned)td0 & 0xFFFFu) << 16);        // tensor_dim0 lo
    g1[2] = (int)((((unsigned)td1 & 0xFFFFu) << 16) |
                  ((unsigned)(td0 >> 16) & 0xFFFFu));
    g1[3] = (int)(((tile_d0 & 0xFFFFu) << 16) |
                  ((unsigned)(td1 >> 16) & 0xFFFFu));
    g1[4] = (int)(((tile_d2 & 0xFFFFu) << 16) | (tile_d1 & 0xFFFFu));
    g1[5] = (int)(unsigned)(s0 & 0xFFFFFFFFull);           // dim0_stride[31:0]
    g1[6] = (int)((((unsigned)(s1 & 0xFFFFull)) << 16) |   // dim1_stride[15:0]
                  ((unsigned)((s0 >> 32) & 0xFFFFull)));   // dim0_stride[47:32]
    g1[7] = (int)(unsigned)((s1 >> 16) & 0xFFFFFFFFull);   // dim1_stride[47:16]
    v4i g2;
    g2[0] = (int)(1u << 20);                               // tensor_dim2 (big)
    g2[1] = 0; g2[2] = 0; g2[3] = 0;
    v4i g3 = {0, 0, 0, 0};
#if defined(__clang_major__) && __clang_major__ >= 23
    v8i g4 = {0, 0, 0, 0, 0, 0, 0, 0};
    __builtin_amdgcn_tensor_load_to_lds(g0, g1, g2, g3, g4, 0);
#else
    __builtin_amdgcn_tensor_load_to_lds(g0, g1, g2, g3, 0);
#endif
}

// ---------------------------------------------------------------------------
// GEMM: C[M,Nout] = A[M,K] * W[K,Nout] + bias,  A bf16 (M x K) row-major,
// Wt bf16 pre-transposed (Nout x K). One wave computes 64(m) x 16(n);
// 4 waves/block cover 64 x 64. MODE is a template param -> branch-free.
// ---------------------------------------------------------------------------
template <int MODE>
__global__ __launch_bounds__(128)
void gemm_wmma(const unsigned short* __restrict__ A,
               const unsigned short* __restrict__ Wt,
               const float* __restrict__ bias,
               void* __restrict__ out,
               int M, int Kdim, int Nout)
{
    const int lane = threadIdx.x & 31;
    const int wave = threadIdx.x >> 5;
    const int n0   = blockIdx.x * 64 + wave * 16;
    const int m0   = blockIdx.y * 64;
    const int ml   = lane & 15;
    const int half = lane >> 4;

    v8f acc[4] = {{}, {}, {}, {}};
    const unsigned short* wrow = Wt + (size_t)(n0 + ml) * Kdim;

    for (int kk = 0; kk < Kdim; kk += 32) {
        Frag bfr;
        #pragma unroll
        for (int v = 0; v < 8; ++v)
            bfr.u[v] = *(const unsigned*)(wrow + kk + ((v >> 2) << 4) + (half << 3) + ((v & 3) << 1));
        #pragma unroll
        for (int s = 0; s < 4; ++s) {
            Frag a;
            const unsigned short* arow =
                A + (size_t)(m0 + s * 16 + ml) * Kdim + kk;
            #pragma unroll
            for (int v = 0; v < 8; ++v) {
                int kl = ((v >> 2) << 4) + (half << 3) + ((v & 3) << 1);
                a.u[v] = *(const unsigned*)(arow + kl);
            }
            acc[s] = __builtin_amdgcn_wmma_f32_16x16x32_bf16(
                         false, a.v, false, bfr.v, (short)0, acc[s],
                         false, false);
        }
    }

    const int   n  = n0 + ml;
    const float bv = bias[n];
    #pragma unroll
    for (int s = 0; s < 4; ++s) {
        #pragma unroll
        for (int r = 0; r < 8; ++r) {
            int   m = m0 + s * 16 + r + (half << 3);
            float c = acc[s][r] + bv;
            if (MODE == MODE_F32) {
                ((float*)out)[(size_t)m * Nout + n] = c;
            } else if (MODE == MODE_GELU_BF16) {
                float g = 0.5f * c * (1.0f + erff(c * 0.70710678118f));
                ((unsigned short*)out)[(size_t)m * Nout + n] =
                    f32_to_bf16_bits(g);
            } else if (MODE == MODE_BF16_VT) {
                int bidx = m >> 10;
                int tok  = m & 1023;
                ((unsigned short*)out)[((size_t)(bidx * Nout + n) << 10) + tok] =
                    f32_to_bf16_bits(c);
            } else {
                ((unsigned short*)out)[(size_t)m * Nout + n] =
                    f32_to_bf16_bits(c);
            }
        }
    }
}

// ---------------------------------------------------------------------------
// Flash attention with per-relation masking. One wave per (b, h, 16 rows).
// K tile (32j x 32dh) and V tile (4rel x 32dh x 32j) are double-buffered
// into LDS by the TDM; compute overlaps the next stage's DMA.
// ---------------------------------------------------------------------------
__global__ __launch_bounds__(32)
void attn_kernel(const unsigned short* __restrict__ qb,
                 const unsigned short* __restrict__ kb,
                 const unsigned short* __restrict__ vbT,
                 const int* __restrict__ adj,
                 unsigned short* __restrict__ aob)
{
    __shared__ unsigned short lds_k[2][32 * 32];        // [buf][j][dh]
    __shared__ unsigned short lds_v[2][4 * 32 * 32];    // [buf][rel][dh][j]
    __shared__ unsigned short lds_p[16 * 32];

    const int lane = threadIdx.x & 31;
    const int ml   = lane & 15;
    const int half = lane >> 4;
    const int i0   = blockIdx.x * 16;
    const int h    = blockIdx.y;
    const int b    = blockIdx.z;
    const float scale = 0.17677669529663687f;   // 1/sqrt(32)

    const unsigned k_lds0 = (unsigned)(size_t)&lds_k[0][0];
    const unsigned v_lds0 = (unsigned)(size_t)&lds_v[0][0];

    // TDM issue for one 32-wide j stage into buffer `buf`
    auto issue_stage = [&](int j0, int buf) {
        unsigned long long ka = (unsigned long long)(size_t)
            (kb + ((size_t)(b * 1024 + j0) << 8) + h * 32);
        tdm_load_bf16(k_lds0 + buf * 2048, ka, 32, 32, 256ull, 0, 0ull);
        unsigned long long va = (unsigned long long)(size_t)
            (vbT + (((size_t)(b * 256 + h * 32)) << 10) + j0);
        tdm_load_bf16(v_lds0 + buf * 8192, va, 32, 32, 1024ull, 4,
                      (1ull << 21));
    };
    issue_stage(0, 0);
    issue_stage(32, 1);

    // Q A-frag: rows i0+ml, K = dh 0..31 (contiguous pairs)
    Frag qf;
    const unsigned short* qrow =
        qb + ((size_t)(b * 1024 + i0 + ml) << 8) + h * 32;
    #pragma unroll
    for (int v = 0; v < 8; ++v) {
        int kl = ((v >> 2) << 4) + (half << 3) + ((v & 3) << 1);
        qf.u[v] = *(const unsigned*)(qrow + kl);
    }

    float mrow[8], lrow[8];
    v8f O0 = {}, O1 = {};
    #pragma unroll
    for (int r = 0; r < 8; ++r) { mrow[r] = -1e30f; lrow[r] = 0.0f; }

    for (int j0 = 0; j0 < 1024; j0 += 32) {
        const int buf = (j0 >> 5) & 1;
        if (j0 + 32 < 1024) __builtin_amdgcn_s_wait_tensorcnt(2);
        else                __builtin_amdgcn_s_wait_tensorcnt(0);

        // compiler barrier: TDM wrote LDS; block hoisting/elimination of the
        // ds reads below while keeping them in addrspace(3) (ds_load path).
        asm volatile("" ::: "memory");
        const unsigned short* kbuf = &lds_k[buf][0];
        const unsigned short* vbuf = &lds_v[buf][0];

        // scores: S[t] = Q(16x32) * K^T(32x16) for two 16-col j tiles
        v8f S[2];
        #pragma unroll
        for (int t = 0; t < 2; ++t) {
            Frag kf;
            const unsigned short* krow =
                kbuf + (t * 16 + ml) * 32;
            #pragma unroll
            for (int v = 0; v < 8; ++v)
                kf.u[v] = *(const unsigned*)(krow + ((v >> 2) << 4) + (half << 3) + ((v & 3) << 1));
            v8f z = {};
            S[t] = __builtin_amdgcn_wmma_f32_16x16x32_bf16(
                       false, qf.v, false, kf.v, (short)0, z, false, false);
        }

        // online softmax (row stats via 16-lane shfl_xor reductions)
        float p0v[8], p1v[8];
        #pragma unroll
        for (int r = 0; r < 8; ++r) {
            float s0 = S[0][r] * scale, s1 = S[1][r] * scale;
            float tm = fmaxf(s0, s1);
            #pragma unroll
            for (int msk = 1; msk < 16; msk <<= 1)
                tm = fmaxf(tm, __shfl_xor(tm, msk, 32));
            float mn   = fmaxf(mrow[r], tm);
            float corr = __expf(mrow[r] - mn);
            float p0   = __expf(s0 - mn);
            float p1   = __expf(s1 - mn);
            float ts   = p0 + p1;
            #pragma unroll
            for (int msk = 1; msk < 16; msk <<= 1)
                ts += __shfl_xor(ts, msk, 32);
            lrow[r] = lrow[r] * corr + ts;
            mrow[r] = mn;
            O0[r] *= corr; O1[r] *= corr;
            p0v[r] = p0;   p1v[r] = p1;
        }

        // stage P (bf16) through LDS: C-layout -> row-major 16x32
        __syncthreads();
        #pragma unroll
        for (int r = 0; r < 8; ++r) {
            int m = r + (half << 3);
            lds_p[m * 32 + ml]      = f32_to_bf16_bits(p0v[r]);
            lds_p[m * 32 + 16 + ml] = f32_to_bf16_bits(p1v[r]);
        }
        __syncthreads();

        // reload P in A-frag layout + adjacency pairs
        unsigned praw[8];
        int2 adjp[8];
        const int* adjrow = adj + ((size_t)b << 20) +
                            (size_t)(i0 + ml) * 1024 + j0;
        #pragma unroll
        for (int v = 0; v < 8; ++v) {
            int jl = ((v >> 2) << 4) + (half << 3) + ((v & 3) << 1);
            praw[v] = *(const unsigned*)&lds_p[ml * 32 + jl];
            adjp[v] = *(const int2*)(adjrow + jl);
        }

        // 4 relations: masked P (A-frag) x V_r (B-frag from LDS)
        #pragma unroll
        for (int rr = 0; rr < 4; ++rr) {
            Frag pf;
            #pragma unroll
            for (int v = 0; v < 8; ++v) {
                unsigned lo = (adjp[v].x == rr) ? (praw[v] & 0x0000FFFFu) : 0u;
                unsigned hi = (adjp[v].y == rr) ? (praw[v] & 0xFFFF0000u) : 0u;
                pf.u[v] = lo | hi;
            }
            #pragma unroll
            for (int t = 0; t < 2; ++t) {
                Frag vf;
                const unsigned short* vrow =
                    vbuf + rr * 1024 + (t * 16 + ml) * 32;
                #pragma unroll
                for (int v = 0; v < 8; ++v)
                    vf.u[v] = *(const unsigned*)(vrow + ((v >> 2) << 4) + (half << 3) + ((v & 3) << 1));
                if (t == 0)
                    O0 = __builtin_amdgcn_wmma_f32_16x16x32_bf16(
                             false, pf.v, false, vf.v, (short)0, O0,
                             false, false);
                else
                    O1 = __builtin_amdgcn_wmma_f32_16x16x32_bf16(
                             false, pf.v, false, vf.v, (short)0, O1,
                             false, false);
            }
        }

        if (j0 + 64 < 1024) issue_stage(j0 + 64, buf);
    }

    // normalize and store bf16 into (B,N,D) for the Wo GEMM
    #pragma unroll
    for (int r = 0; r < 8; ++r) {
        float inv = 1.0f / lrow[r];
        int   m   = i0 + r + (half << 3);
        size_t base = ((size_t)(b * 1024 + m) << 8) + h * 32;
        aob[base + ml]      = f32_to_bf16_bits(O0[r] * inv);
        aob[base + 16 + ml] = f32_to_bf16_bits(O1[r] * inv);
    }
}

// ---------------------------------------------------------------------------
// Elementwise helpers
// ---------------------------------------------------------------------------
__global__ void cvt_f32_bf16(const float* __restrict__ in,
                             unsigned short* __restrict__ out, int n)
{
    int i = blockIdx.x * blockDim.x + threadIdx.x;
    if (i < n) out[i] = f32_to_bf16_bits(in[i]);
}

// Wt[n*K + k] = bf16(W[k*N + n])  (one-time weight transpose+convert)
__global__ void transpose_cvt(const float* __restrict__ W,
                              unsigned short* __restrict__ Wt, int K, int N)
{
    int idx = blockIdx.x * blockDim.x + threadIdx.x;
    if (idx >= K * N) return;
    int k = idx / N;
    int n = idx - k * N;
    Wt[(size_t)n * K + k] = f32_to_bf16_bits(W[idx]);
}

// x = LN(a + b) over D=256; optional bf16 mirror for the next GEMM
__global__ __launch_bounds__(256)
void add_ln(const float* __restrict__ a, const float* __restrict__ bsrc,
            const float* __restrict__ g, const float* __restrict__ beta,
            float* __restrict__ xout, unsigned short* __restrict__ xb)
{
    __shared__ float red[256];
    const int row = blockIdx.x;
    const int t   = threadIdx.x;
    const size_t idx = ((size_t)row << 8) + t;
    float v = a[idx] + bsrc[idx];

    red[t] = v; __syncthreads();
    for (int s = 128; s > 0; s >>= 1) {
        if (t < s) red[t] += red[t + s];
        __syncthreads();
    }
    float mean = red[0] * (1.0f / 256.0f);
    __syncthreads();
    float d = v - mean;
    red[t] = d * d; __syncthreads();
    for (int s = 128; s > 0; s >>= 1) {
        if (t < s) red[t] += red[t + s];
        __syncthreads();
    }
    float var = red[0] * (1.0f / 256.0f);
    float y = d * rsqrtf(var + 1e-5f) * g[t] + beta[t];
    xout[idx] = y;
    if (xb) xb[idx] = f32_to_bf16_bits(y);
}

// ---------------------------------------------------------------------------
extern "C" void kernel_launch(void* const* d_in, const int* in_sizes, int n_in,
                              void* d_out, int out_size, void* d_ws, size_t ws_size,
                              hipStream_t stream)
{
    const float* src   = (const float*)d_in[0];
    const int*   adj   = (const int*)  d_in[1];
    const float* Wq    = (const float*)d_in[2];
    const float* bq    = (const float*)d_in[3];
    const float* Wk    = (const float*)d_in[4];
    const float* bk    = (const float*)d_in[5];
    const float* Wv    = (const float*)d_in[6];   // (4,256,256)
    const float* bv    = (const float*)d_in[7];   // (4,256)
    const float* Wo    = (const float*)d_in[8];
    const float* bo    = (const float*)d_in[9];
    const float* W1    = (const float*)d_in[10];
    const float* b1    = (const float*)d_in[11];
    const float* W2    = (const float*)d_in[12];
    const float* b2    = (const float*)d_in[13];
    const float* g1    = (const float*)d_in[14];
    const float* beta1 = (const float*)d_in[15];
    const float* g2    = (const float*)d_in[16];
    const float* beta2 = (const float*)d_in[17];

    const int BN = 8 * 1024;          // 8192 token rows
    const int D  = 256, FF = 1024;

    // workspace carve-up (256B aligned)
    char* ws = (char*)d_ws;
    size_t off = 0;
    auto carve = [&](size_t bytes) {
        void* p = ws + off;
        off += (bytes + 255) & ~(size_t)255;
        return p;
    };
    unsigned short* srcb = (unsigned short*)carve((size_t)BN * D * 2);
    unsigned short* qb   = (unsigned short*)carve((size_t)BN * D * 2);
    unsigned short* kb   = (unsigned short*)carve((size_t)BN * D * 2);
    unsigned short* vbT  = (unsigned short*)carve((size_t)4 * BN * D * 2);
    unsigned short* aob  = (unsigned short*)carve((size_t)BN * D * 2);
    float*          yo   = (float*)         carve((size_t)BN * D * 4);
    float*          x    = (float*)         carve((size_t)BN * D * 4);
    unsigned short* xb   = (unsigned short*)carve((size_t)BN * D * 2);
    unsigned short* hb   = (unsigned short*)carve((size_t)BN * FF * 2);
    float*          ff   = (float*)         carve((size_t)BN * D * 4);
    unsigned short* wqT  = (unsigned short*)carve((size_t)D * D * 2);
    unsigned short* wkT  = (unsigned short*)carve((size_t)D * D * 2);
    unsigned short* wvT  = (unsigned short*)carve((size_t)4 * D * D * 2);
    unsigned short* woT  = (unsigned short*)carve((size_t)D * D * 2);
    unsigned short* w1T  = (unsigned short*)carve((size_t)D * FF * 2);
    unsigned short* w2T  = (unsigned short*)carve((size_t)FF * D * 2);

    // 0) one-time conversions
    cvt_f32_bf16<<<(BN * D + 255) / 256, 256, 0, stream>>>(src, srcb, BN * D);
    transpose_cvt<<<(D * D + 255) / 256, 256, 0, stream>>>(Wq, wqT, D, D);
    transpose_cvt<<<(D * D + 255) / 256, 256, 0, stream>>>(Wk, wkT, D, D);
    for (int r = 0; r < 4; ++r)
        transpose_cvt<<<(D * D + 255) / 256, 256, 0, stream>>>(
            Wv + (size_t)r * D * D, wvT + (size_t)r * D * D, D, D);
    transpose_cvt<<<(D * D + 255) / 256, 256, 0, stream>>>(Wo, woT, D, D);
    transpose_cvt<<<(D * FF + 255) / 256, 256, 0, stream>>>(W1, w1T, D, FF);
    transpose_cvt<<<(FF * D + 255) / 256, 256, 0, stream>>>(W2, w2T, FF, D);

    // 1) projections (WMMA, 64x64 tile per 4-wave block)
    dim3 gD(D / 64, BN / 64);
    gemm_wmma<MODE_BF16><<<gD, 128, 0, stream>>>(srcb, wqT, bq, qb, BN, D, D);
    gemm_wmma<MODE_BF16><<<gD, 128, 0, stream>>>(srcb, wkT, bk, kb, BN, D, D);
    for (int r = 0; r < 4; ++r) {
        gemm_wmma<MODE_BF16_VT><<<gD, 128, 0, stream>>>(
            srcb, wvT + (size_t)r * D * D, bv + (size_t)r * D,
            vbT + ((size_t)r << 21), BN, D, D);
    }

    // 2) flash attention with relation masking (TDM-staged K/V)
    attn_kernel<<<dim3(64, 8, 8), 32, 0, stream>>>(qb, kb, vbT, adj, aob);

    // 3) output projection + first residual LN
    gemm_wmma<MODE_F32><<<gD, 128, 0, stream>>>(aob, woT, bo, yo, BN, D, D);
    add_ln<<<BN, 256, 0, stream>>>(src, yo, g1, beta1, x, xb);

    // 4) FFN: gelu(x@W1+b1)@W2+b2, then final residual LN -> d_out
    dim3 gFF(FF / 64, BN / 64);
    gemm_wmma<MODE_GELU_BF16><<<gFF, 128, 0, stream>>>(xb, w1T, b1, hb, BN, D, FF);
    gemm_wmma<MODE_F32><<<gD, 128, 0, stream>>>(hb, w2T, b2, ff, BN, FF, D);
    add_ln<<<BN, 256, 0, stream>>>(x, ff, g2, beta2, (float*)d_out, nullptr);
}
